// RqVae_22067541967744
// MI455X (gfx1250) — compile-verified, matrix operand order checked by reading, not run on hardware
//
#include <hip/hip_runtime.h>
#include <hip/hip_bf16.h>

// Problem sizes (fixed by the reference).
#define B_  8192
#define K_  8192
#define D_  512

// Tiling: 4 waves/block, each wave owns 16 rows of x (A-stationary in VGPRs).
#define BM  64            // x rows per block (4 waves x 16)
#define BN  16            // codebook rows per K-tile (one 16x16 WMMA column tile)
#define NTHREADS 128
#define LDS_STRIDE 520    // bf16 elems per row: 512 + 8 pad (TDM pad: 16B per 1024B row)

typedef __attribute__((ext_vector_type(16))) __bf16       v16bf;
typedef __attribute__((ext_vector_type(8)))  float        v8f;
typedef __attribute__((ext_vector_type(4)))  unsigned int u32x4;
typedef __attribute__((ext_vector_type(8)))  unsigned int u32x8;
typedef __attribute__((ext_vector_type(8)))  int          i32x8;
typedef __attribute__((ext_vector_type(4)))  int          i32x4;
typedef __attribute__((ext_vector_type(4)))  float        f32x4;

// ---------------------------------------------------------------------------
// TDM: issue one tensor_load_to_lds for a 16x512 bf16 tile at gsrc -> lds_addr.
// D# group0: count=1, lds_addr, 57b global addr, type=2 ("image").
// D# group1: data_size=2B, pad_enable, pad_interval=7 (256 DW = 1 row),
//            pad_amount=3 (4 DW = 16B)  => LDS row stride 520 bf16.
//            tensor_dim0=512, tensor_dim1=K, tile_dim0=512, tile_dim1=16,
//            tensor_dim0_stride=512 elements.
// This toolchain exposes the 6-arg builtin:
//   (u32x4 g0, i32x8 g1, i32x4 g2, i32x4 g3, i32x8 extra, i32 cpol)
// ---------------------------------------------------------------------------
__device__ __forceinline__ void tdm_load_tile(const __bf16* gsrc, unsigned lds_addr) {
    unsigned long long ga = (unsigned long long)(uintptr_t)gsrc;
    u32x4 g0;
    g0[0] = 1u;                                   // count=1 (valid), no gather
    g0[1] = lds_addr;                             // LDS byte address
    g0[2] = (unsigned)(ga & 0xFFFFFFFFu);         // global_addr[31:0]
    g0[3] = (unsigned)((ga >> 32) & 0x01FFFFFFu)  // global_addr[56:32]
            | (2u << 30);                         // type=2
    i32x8 g1;
    g1[0] = (1 << 16)                             // data_size = 2 bytes
          | (1 << 20)                             // pad_enable
          | (7 << 22)                             // pad_interval: 256 DWORDs
          | (3 << 25);                            // pad_amount: 4 DWORDs (16B)
    g1[1] = (int)(512u << 16);                    // tensor_dim0[15:0]=512
    g1[2] = (int)((unsigned)K_ << 16);            // tensor_dim0 hi=0 | tensor_dim1 lo
    g1[3] = (int)(512u << 16);                    // tensor_dim1 hi=0 | tile_dim0=512
    g1[4] = BN;                                   // tile_dim1=16, tile_dim2=0
    g1[5] = 512;                                  // tensor_dim0_stride = 512 elems
    g1[6] = 0;                                    // stride hi | tensor_dim1_stride lo
    g1[7] = 0;
    i32x4 z4 = {0, 0, 0, 0};                      // groups 2/3: dims beyond 2D unused
    i32x8 z8 = {0, 0, 0, 0, 0, 0, 0, 0};          // extra group (zero-filled)
    __builtin_amdgcn_tensor_load_to_lds(g0, g1, z4, z4, z8, 0 /*cpol*/);
}

// ---------------------------------------------------------------------------
// Kernel 1: codebook -> bf16 copy + per-row squared norms.
// ---------------------------------------------------------------------------
__global__ void __launch_bounds__(256)
prep_codebook(const float* __restrict__ e,
              __bf16* __restrict__ eb,
              float* __restrict__ esq) {
    const int lane = threadIdx.x & 31;
    const int wave = threadIdx.x >> 5;
    const int row  = blockIdx.x * 8 + wave;

    const float* src = e  + (size_t)row * D_;
    __bf16*      dst = eb + (size_t)row * D_;

    float s = 0.0f;
#pragma unroll
    for (int i = 0; i < D_ / 32; ++i) {
        float v = src[lane + i * 32];
        s += v * v;
        dst[lane + i * 32] = (__bf16)v;
    }
#pragma unroll
    for (int m = 16; m >= 1; m >>= 1)
        s += __shfl_xor(s, m, 32);
    if (lane == 0) esq[row] = s;
}

// ---------------------------------------------------------------------------
// Kernel 2: fused cross-GEMM (bf16 WMMA) + argmin + residual.
// TDM double-buffers the codebook tile into LDS, overlapped with compute.
// ---------------------------------------------------------------------------
__global__ void __launch_bounds__(NTHREADS)
vq_argmin_kernel(const float* __restrict__ x,
                 const float* __restrict__ e,
                 const __bf16* __restrict__ eb,
                 const float* __restrict__ esq,
                 int* __restrict__ idx_out,
                 float* __restrict__ res_out) {
    __shared__ __align__(16) __bf16 es[2][BN * LDS_STRIDE];  // 2 x 16.6 KB
    __shared__ int rowIdx[BM];

    const int tid     = threadIdx.x;
    const int lane    = tid & 31;
    const int wave    = tid >> 5;        // 0..3
    const int half    = lane >> 4;       // 0 | 1
    const int l16     = lane & 15;
    const int rowBase = blockIdx.x * BM;
    const int waveRow = rowBase + wave * 16;

    // ---- Preload A fragments: 16 chunks of 16x32 (bf16), ISA 7.12.2 layout.
    v16bf afrag[16];
    {
        const float* xr = x + (size_t)(waveRow + l16) * D_;
#pragma unroll
        for (int c = 0; c < 16; ++c) {
            const int d0 = c * 32 + half * 8;
            f32x4 q0 = *(const f32x4*)(xr + d0);
            f32x4 q1 = *(const f32x4*)(xr + d0 + 4);
            f32x4 q2 = *(const f32x4*)(xr + d0 + 16);
            f32x4 q3 = *(const f32x4*)(xr + d0 + 20);
            v16bf a;
#pragma unroll
            for (int i = 0; i < 4; ++i) {
                a[i]      = (__bf16)q0[i];
                a[4 + i]  = (__bf16)q1[i];
                a[8 + i]  = (__bf16)q2[i];
                a[12 + i] = (__bf16)q3[i];
            }
            afrag[c] = a;
        }
    }

    float best[8];
    int   bidx[8];
#pragma unroll
    for (int r = 0; r < 8; ++r) { best[r] = 3.4e38f; bidx[r] = 0; }

    const unsigned ldsBuf0 = (unsigned)(uintptr_t)&es[0][0];
    const unsigned ldsBuf1 = (unsigned)(uintptr_t)&es[1][0];

    // Prologue: kick off tile 0 (wave 0 only; real branch, TDM ignores EXEC).
    if (wave == 0) {
        tdm_load_tile(eb, ldsBuf0);
    }

    const int NT = K_ / BN;   // 512 K-tiles
    for (int t = 0; t < NT; ++t) {
        if (wave == 0) {
            __builtin_amdgcn_s_wait_tensorcnt(0);   // tile t landed in LDS
        }
        __syncthreads();  // tile t visible to all; compute t-1 done -> buf free
        if (wave == 0 && (t + 1) < NT) {
            tdm_load_tile(eb + (size_t)(t + 1) * BN * D_,
                          ((t + 1) & 1) ? ldsBuf1 : ldsBuf0);  // overlaps compute
        }

        const __bf16* esb = &es[t & 1][0];

        // ---- cross tile: 16 WMMAs over D = 512 (A resident in VGPRs).
        v8f acc = {0.f, 0.f, 0.f, 0.f, 0.f, 0.f, 0.f, 0.f};
#pragma unroll
        for (int c = 0; c < 16; ++c) {
            const int d0   = c * 32 + half * 8;
            const int base = l16 * LDS_STRIDE + d0;   // B col n = l16, K-major
            u32x4 lo = *(const u32x4*)&esb[base];
            u32x4 hi = *(const u32x4*)&esb[base + 16];
            u32x8 raw = __builtin_shufflevector(lo, hi, 0, 1, 2, 3, 4, 5, 6, 7);
            v16bf b = __builtin_bit_cast(v16bf, raw);
            acc = __builtin_amdgcn_wmma_f32_16x16x32_bf16(
                false, afrag[c], false, b, (short)0, acc, false, false);
        }

        // ---- fold distances into running argmin (lane column = kb + l16).
        const int   n  = t * BN + l16;
        const float eq = esq[n];
#pragma unroll
        for (int r = 0; r < 8; ++r) {
            float d = eq - 2.0f * acc[r];
            if (d < best[r]) { best[r] = d; bidx[r] = n; }
        }
    }

    // ---- Reduce the 16 columns held by each 16-lane half (rows = 8*half + r).
#pragma unroll
    for (int r = 0; r < 8; ++r) {
        float bv = best[r];
        int   bi = bidx[r];
#pragma unroll
        for (int m = 1; m <= 8; m <<= 1) {
            float ov = __shfl_xor(bv, m, 32);
            int   oi = __shfl_xor(bi, m, 32);
            if (ov < bv || (ov == bv && oi < bi)) { bv = ov; bi = oi; }
        }
        best[r] = bv;
        bidx[r] = bi;
    }
    if (l16 == 0) {
#pragma unroll
        for (int r = 0; r < 8; ++r) {
            int row = wave * 16 + half * 8 + r;
            rowIdx[row] = bidx[r];
            idx_out[rowBase + row] = bidx[r];
        }
    }
    __syncthreads();

    // ---- Residual in full fp32: out = x - e[argmin].
    {
        const f32x4* xv = (const f32x4*)(x + (size_t)rowBase * D_);
        f32x4*       ov = (f32x4*)(res_out + (size_t)rowBase * D_);
        for (int j = tid; j < BM * (D_ / 4); j += NTHREADS) {
            int row   = j >> 7;                     // 128 f32x4 per row
            int col   = j & 127;
            int kbest = rowIdx[row];
            const f32x4* ev = (const f32x4*)(e + (size_t)kbest * D_);
            ov[j] = xv[j] - ev[col];
        }
    }
}

// ---------------------------------------------------------------------------
extern "C" void kernel_launch(void* const* d_in, const int* in_sizes, int n_in,
                              void* d_out, int out_size, void* d_ws, size_t ws_size,
                              hipStream_t stream) {
    (void)in_sizes; (void)n_in; (void)out_size; (void)ws_size;

    const float* x = (const float*)d_in[0];   // previous_residual [B, D] fp32
    const float* e = (const float*)d_in[1];   // codebook_embeddings [K, D] fp32

    // Workspace layout: e_sq [K] fp32, then bf16 codebook [K, D].
    float*  esq = (float*)d_ws;
    __bf16* eb  = (__bf16*)((char*)d_ws + (size_t)K_ * sizeof(float));

    // Output layout: indices [B] int32, then next_residual [B, D] fp32.
    int*   idx_out = (int*)d_out;
    float* res_out = (float*)((char*)d_out + (size_t)B_ * sizeof(int));

    prep_codebook<<<K_ / 8, 256, 0, stream>>>(e, eb, esq);
    vq_argmin_kernel<<<B_ / BM, NTHREADS, 0, stream>>>(x, e, eb, esq, idx_out, res_out);
}
